// RQVAE_56384330662368
// MI455X (gfx1250) — matrix operand order repaired and checked by reading, not compiled, not run on hardware
//
#include <hip/hip_runtime.h>

// ---------------------------------------------------------------------------
// RQ-VAE forward for MI455X (gfx1250, wave32, WMMA bf16 16x16x32, TDM)
// bf16 end-to-end activation dataflow; 16x64 register-blocked WMMA GEMM.
// ---------------------------------------------------------------------------

#define B_SZ    32768
#define D_IN    512
#define HID     256
#define D_CODE  128
#define KCB     8192
#define LVLS    4
#define BETA    0.25f

typedef __attribute__((ext_vector_type(16))) __bf16 v16bf;
typedef __attribute__((ext_vector_type(8)))  float  v8f;
typedef __attribute__((ext_vector_type(4)))  unsigned int v4u32;
typedef __attribute__((ext_vector_type(8)))  int    v8i32;
typedef __attribute__((ext_vector_type(4)))  int    v4i32;

#if defined(__gfx1250__) && __has_builtin(__builtin_amdgcn_tensor_load_to_lds) && __has_builtin(__builtin_amdgcn_s_wait_tensorcnt)
#define USE_TDM 1
#else
#define USE_TDM 0
#endif

union Frag {
    unsigned u[8];
    v16bf    v;
};

static __device__ __forceinline__ unsigned pack2bf(float a, float b) {
    unsigned ua = __float_as_uint(a), ub = __float_as_uint(b);
    ua = (ua + 0x7FFFu + ((ua >> 16) & 1u)) >> 16;
    ub = (ub + 0x7FFFu + ((ub >> 16) & 1u)) >> 16;
    return (ub << 16) | (ua & 0xFFFFu);
}

static __device__ __forceinline__ unsigned short f2bf(float a) {
    unsigned ua = __float_as_uint(a);
    return (unsigned short)((ua + 0x7FFFu + ((ua >> 16) & 1u)) >> 16);
}

// ---------------------------------------------------------------------------
// fp32 -> bf16 bulk convert (8 elements / thread)
// ---------------------------------------------------------------------------
__global__ __launch_bounds__(256) void f32_to_bf16(const float* __restrict__ src,
                                                   unsigned short* __restrict__ dst,
                                                   long n) {
    long i = ((long)blockIdx.x * 256 + threadIdx.x) * 8;
    if (i >= n) return;
    const float4* p = (const float4*)(src + i);
    float4 f0 = p[0], f1 = p[1];
    uint4 o;
    o.x = pack2bf(f0.x, f0.y); o.y = pack2bf(f0.z, f0.w);
    o.z = pack2bf(f1.x, f1.y); o.w = pack2bf(f1.z, f1.w);
    *(uint4*)(dst + i) = o;
}

// ---------------------------------------------------------------------------
// Weight prep: W (K x N) fp32 row-major -> Wt (N x K) bf16 (column of W contig)
// ---------------------------------------------------------------------------
__global__ __launch_bounds__(256) void wt_prep(const float* __restrict__ W,
                                               unsigned short* __restrict__ Wt,
                                               int K, int N) {
    int i = blockIdx.x * blockDim.x + threadIdx.x;
    if (i >= N * K) return;
    int n = i / K, k = i % K;
    Wt[i] = f2bf(W[(size_t)k * N + n]);
}

// ---------------------------------------------------------------------------
// Codebook prep: fp32 -> bf16 copy + per-code squared norm
// ---------------------------------------------------------------------------
__global__ __launch_bounds__(256) void cb_prep(const float* __restrict__ cb,
                                               unsigned short* __restrict__ cbbf,
                                               float* __restrict__ cbnorm) {
    int i = blockIdx.x * blockDim.x + threadIdx.x;   // over LVLS*KCB codes
    if (i >= LVLS * KCB) return;
    const float* src = cb + (size_t)i * D_CODE;
    unsigned short* dst = cbbf + (size_t)i * D_CODE;
    float s = 0.f;
    for (int k = 0; k < D_CODE; k++) {
        float v = src[k];
        s += v * v;
        dst[k] = f2bf(v);
    }
    cbnorm[i] = s;
}

// ---------------------------------------------------------------------------
// WMMA GEMM: C(M,N) = act(A(M,K) @ W(K,N) + bias); A bf16 row-major,
// W transposed bf16 (N x K). One wave computes a 16x64 tile: one A-fragment
// amortized across 4 B-fragments / 4 WMMAs per K-step (zero conversion VALU
// in the K-loop). Optional f32 and/or bf16 outputs.
// A-frag (16-bit 16x32): lane m=l%16; K 0-15 (lanes<16),
//                        K 16-31 (lanes>=16), 2 packed / VGPR.
// B-frag (16-bit 32x16): lane n=l%16; K 0-15 (lanes<16), 16-31 (>=16).
// C layout: VGPR j -> row j (+8 for lanes>=16), col = l%16.
// ---------------------------------------------------------------------------
__global__ __launch_bounds__(128) void wmma_gemm_bias(
        const unsigned short* __restrict__ A, const unsigned short* __restrict__ Bt,
        const float* __restrict__ bias, float* __restrict__ Cf32,
        unsigned short* __restrict__ Cbf, int M, int K, int N, int relu) {
    const int lane = threadIdx.x & 31;
    const int wave = threadIdx.x >> 5;
    const int tile = blockIdx.x * 4 + wave;     // 16x64 tiles
    const int tn = N >> 6;                      // col-tiles of 64
    const int tm = tile / tn;
    const int tc = tile - tm * tn;
    if (tm >= (M >> 4)) return;
    const int row0 = tm << 4, col0 = tc << 6;
    const int r = row0 + (lane & 15);

    v8f acc[4];
#pragma unroll
    for (int i = 0; i < 4; i++) {
        const float bv = bias[col0 + i * 16 + (lane & 15)];
#pragma unroll
        for (int j = 0; j < 8; j++) acc[i][j] = bv;
    }

    for (int k = 0; k < K; k += 32) {
        Frag a;
        const int ka = k + ((lane & 16) ? 16 : 0);
        uint4 a0 = *(const uint4*)(A + (size_t)r * K + ka);
        uint4 a1 = *(const uint4*)(A + (size_t)r * K + ka + 8);
        a.u[0] = a0.x; a.u[1] = a0.y; a.u[2] = a0.z; a.u[3] = a0.w;
        a.u[4] = a1.x; a.u[5] = a1.y; a.u[6] = a1.z; a.u[7] = a1.w;

        const int kb = k + ((lane & 16) ? 16 : 0);
#pragma unroll
        for (int i = 0; i < 4; i++) {
            Frag b;
            const int c = col0 + i * 16 + (lane & 15);
            const uint4* pb = (const uint4*)(Bt + (size_t)c * K + kb);
            uint4 b0 = pb[0], b1 = pb[1];
            b.u[0] = b0.x; b.u[1] = b0.y; b.u[2] = b0.z; b.u[3] = b0.w;
            b.u[4] = b1.x; b.u[5] = b1.y; b.u[6] = b1.z; b.u[7] = b1.w;
            acc[i] = __builtin_amdgcn_wmma_f32_16x16x32_bf16(
                    false, a.v, false, b.v, (short)0, acc[i], false, false);
        }
    }

#pragma unroll
    for (int i = 0; i < 4; i++) {
        const int c = col0 + i * 16 + (lane & 15);
#pragma unroll
        for (int j = 0; j < 8; j++) {
            int m = row0 + j + ((lane & 16) ? 8 : 0);
            float v = acc[i][j];
            if (relu) v = fmaxf(v, 0.0f);
            if (Cf32) Cf32[(size_t)m * N + c] = v;
            if (Cbf)  Cbf[(size_t)m * N + c] = f2bf(v);
        }
    }
}

// ---------------------------------------------------------------------------
// Residual quantization: each wave owns 16 rows. Per level:
//   argmin_k ( ||cb_k||^2 - 2 * r . cb_k )  via WMMA over all 8192 codes,
//   gather fp32 code vector, update residual, accumulate (q-r)^2.
// Codebook tiles (64 codes x 128 dims bf16 = 16KB) streamed by the TDM.
// ---------------------------------------------------------------------------
__global__ __launch_bounds__(128) void rq_quant(
        const float* __restrict__ ze, const float* __restrict__ cb32,
        const unsigned short* __restrict__ cbbf,
        const float* __restrict__ cbnorm,
        float* __restrict__ zq_out, unsigned short* __restrict__ zq_bf,
        float* __restrict__ codes_out, float* __restrict__ partial_cb) {
    __shared__ __align__(16) float res[4][16][D_CODE];      // 32 KB
    __shared__ int idxs[4][16];
#if USE_TDM
    __shared__ __align__(16) unsigned short cbtile[4][64 * D_CODE]; // 64 KB
#endif
    const int lane = threadIdx.x & 31;
    const int w    = threadIdx.x >> 5;
    const int rg   = blockIdx.x * 4 + w;           // 0..2047
    const int row0 = rg << 4;

    // cooperative load: residual = z_e (16 x 128 fp32, 512 float4 / wave)
    {
        const float4* zr = (const float4*)(ze + (size_t)row0 * D_CODE);
        float4* rd = (float4*)&res[w][0][0];
        for (int t = 0; t < 16; t++) rd[lane * 16 + t] = zr[lane * 16 + t];
    }

    float lossAcc = 0.f;

    for (int lvl = 0; lvl < LVLS; lvl++) {
        __syncthreads();   // residual tile stable

        // hoist A-fragments for K=128 (4 chunks of 32) from LDS residual
        Frag af[4];
        const int m = lane & 15;
#pragma unroll
        for (int kk = 0; kk < 4; kk++) {
            const int ka = kk * 32 + ((lane & 16) ? 16 : 0);
            const float4* pr  = (const float4*)&res[w][m][ka];
            const float4* pr2 = (const float4*)&res[w][m][ka + 8];
            float4 f0 = pr[0], f1 = pr[1], f2 = pr2[0], f3 = pr2[1];
            af[kk].u[0] = pack2bf(f0.x, f0.y); af[kk].u[1] = pack2bf(f0.z, f0.w);
            af[kk].u[2] = pack2bf(f1.x, f1.y); af[kk].u[3] = pack2bf(f1.z, f1.w);
            af[kk].u[4] = pack2bf(f2.x, f2.y); af[kk].u[5] = pack2bf(f2.z, f2.w);
            af[kk].u[6] = pack2bf(f3.x, f3.y); af[kk].u[7] = pack2bf(f3.z, f3.w);
        }

        float minv[8];
        int   mini[8];
#pragma unroll
        for (int j = 0; j < 8; j++) { minv[j] = 3.4e38f; mini[j] = 0; }

        const unsigned short* cbL = cbbf + (size_t)lvl * KCB * D_CODE;
        const float* cnL = cbnorm + (size_t)lvl * KCB;

        for (int ctb = 0; ctb < KCB / 64; ctb++) {      // 128 chunks of 64 codes
#if USE_TDM
            {   // TDM: 64x128 bf16 tile -> LDS (D# per CDNA5 ISA)
                unsigned long long ga =
                    (unsigned long long)(const void*)(cbL + (size_t)ctb * 64 * D_CODE);
                unsigned lds = (unsigned)(unsigned long long)&cbtile[w][0];
                v4u32 g0;
                g0[0] = 1u;                                   // count=1 valid
                g0[1] = lds;                                  // lds_addr
                g0[2] = (unsigned)ga;                         // global_addr lo
                g0[3] = ((unsigned)(ga >> 32) & 0x01FFFFFFu) | 0x80000000u; // hi | type=2
                v8i32 g1;
                g1[0] = 0x10000;            // wg_mask=0, data_size=1 (2B)
                g1[1] = (int)(128u << 16);  // tensor_dim0 = 128 (low 16 in [63:48])
                g1[2] = (int)(8192u << 16); // tensor_dim1 = 8192 (low 16 in [95:80])
                g1[3] = (int)(128u << 16);  // tile_dim0 = 128 in [127:112]
                g1[4] = 64;                 // tile_dim1 = 64 in [143:128]
                g1[5] = 128;                // tensor_dim0_stride lo = 128
                g1[6] = 0;
                g1[7] = 0;
                v4i32 g2 = {0, 0, 0, 0}, g3 = {0, 0, 0, 0};
#if __clang_major__ >= 23
                v8i32 g4 = {0, 0, 0, 0, 0, 0, 0, 0};
                __builtin_amdgcn_tensor_load_to_lds(g0, g1, g2, g3, g4, 0);
#else
                __builtin_amdgcn_tensor_load_to_lds(g0, g1, g2, g3, 0);
#endif
                __builtin_amdgcn_s_wait_tensorcnt(0);
            }
#endif
            for (int cti = 0; cti < 4; cti++) {
                const int c = (ctb * 4 + cti) * 16 + (lane & 15);
                const float cn = cnL[c];
                v8f acc = {0.f, 0.f, 0.f, 0.f, 0.f, 0.f, 0.f, 0.f};
#pragma unroll
                for (int kk = 0; kk < 4; kk++) {
                    Frag b;
                    const int kb = kk * 32 + ((lane & 16) ? 16 : 0);
#if USE_TDM
                    const uint4* pb = (const uint4*)
                        &cbtile[w][(size_t)(cti * 16 + (lane & 15)) * D_CODE + kb];
#else
                    const uint4* pb = (const uint4*)(cbL + (size_t)c * D_CODE + kb);
#endif
                    uint4 b0 = pb[0], b1 = pb[1];
                    b.u[0] = b0.x; b.u[1] = b0.y; b.u[2] = b0.z; b.u[3] = b0.w;
                    b.u[4] = b1.x; b.u[5] = b1.y; b.u[6] = b1.z; b.u[7] = b1.w;
                    acc = __builtin_amdgcn_wmma_f32_16x16x32_bf16(
                            false, af[kk].v, false, b.v, (short)0, acc, false, false);
                }
#pragma unroll
                for (int j = 0; j < 8; j++) {
                    float cand = cn - 2.0f * acc[j];
                    if (cand < minv[j]) { minv[j] = cand; mini[j] = c; }
                }
            }
        }

        // cross-lane argmin within each 16-lane half (N striped across lanes)
#pragma unroll
        for (int j = 0; j < 8; j++) {
            float v = minv[j];
            int   ix = mini[j];
            for (int off = 1; off < 16; off <<= 1) {
                float ov = __shfl_xor(v, off, 32);
                int   oi = __shfl_xor(ix, off, 32);
                if (ov < v || (ov == v && oi < ix)) { v = ov; ix = oi; }
            }
            minv[j] = v; mini[j] = ix;
        }
        if (lane == 0) {
#pragma unroll
            for (int j = 0; j < 8; j++) idxs[w][j] = mini[j];
        }
        if (lane == 16) {
#pragma unroll
            for (int j = 0; j < 8; j++) idxs[w][8 + j] = mini[j];
        }
        __syncthreads();

        if (lane < 16)
            codes_out[(size_t)(row0 + lane) * LVLS + lvl] = (float)idxs[w][lane];

        // gather fp32 code vector, loss, residual update (disjoint elements)
        const float* cbF = cb32 + (size_t)lvl * KCB * D_CODE;
        for (int t = 0; t < 64; t++) {
            int e = lane * 64 + t;
            int mm = e >> 7, k = e & 127;
            float q = cbF[(size_t)idxs[w][mm] * D_CODE + k];
            float r = res[w][mm][k];
            float d = q - r;
            lossAcc += d * d;
            res[w][mm][k] = r - q;
        }
    }
    __syncthreads();

    // z_q = z_e - residual_final  (f32 output + bf16 copy for decoder GEMM)
    {
        const float* zr = ze + (size_t)row0 * D_CODE;
        float* zo = zq_out + (size_t)row0 * D_CODE;
        unsigned short* zb = zq_bf + (size_t)row0 * D_CODE;
        const float* rd = &res[w][0][0];
        for (int t = 0; t < 64; t++) {
            int e = lane * 64 + t;
            float v = zr[e] - rd[e];
            zo[e] = v;
            zb[e] = f2bf(v);
        }
    }

    for (int off = 1; off < 32; off <<= 1) lossAcc += __shfl_xor(lossAcc, off, 32);
    if (lane == 0) partial_cb[rg] = lossAcc;
}

// ---------------------------------------------------------------------------
// Reconstruction loss partials (deterministic two-stage reduction)
// ---------------------------------------------------------------------------
__global__ __launch_bounds__(256) void recon_loss_partial(
        const float* __restrict__ recon, const float* __restrict__ x,
        float* __restrict__ partial) {
    __shared__ float sm[256];
    float acc = 0.f;
    const size_t N = (size_t)B_SZ * D_IN;
    for (size_t i = (size_t)blockIdx.x * 256 + threadIdx.x; i < N;
         i += (size_t)gridDim.x * 256) {
        float d = recon[i] - x[i];
        acc += d * d;
    }
    sm[threadIdx.x] = acc;
    __syncthreads();
    for (int s = 128; s > 0; s >>= 1) {
        if ((int)threadIdx.x < s) sm[threadIdx.x] += sm[threadIdx.x + s];
        __syncthreads();
    }
    if (threadIdx.x == 0) partial[blockIdx.x] = sm[0];
}

__global__ __launch_bounds__(256) void finalize_losses(
        const float* __restrict__ pc, const float* __restrict__ pr,
        float* __restrict__ out_scalars) {
    __shared__ float s1[256], s2[256];
    float a = 0.f, b = 0.f;
    for (int i = threadIdx.x; i < 2048; i += 256) { a += pc[i]; b += pr[i]; }
    s1[threadIdx.x] = a; s2[threadIdx.x] = b;
    __syncthreads();
    for (int s = 128; s > 0; s >>= 1) {
        if ((int)threadIdx.x < s) {
            s1[threadIdx.x] += s1[threadIdx.x + s];
            s2[threadIdx.x] += s2[threadIdx.x + s];
        }
        __syncthreads();
    }
    if (threadIdx.x == 0) {
        float cb = s1[0] / ((float)B_SZ * (float)D_CODE);
        float rc = s2[0] / ((float)B_SZ * (float)D_IN);
        out_scalars[0] = cb;                       // cb_total
        out_scalars[1] = BETA * cb;                // beta * commit (== cb fwd)
        out_scalars[2] = rc;                       // recon_loss
        out_scalars[3] = cb + BETA * cb + rc;      // total
    }
}

// ---------------------------------------------------------------------------
// Host launcher
// ---------------------------------------------------------------------------
extern "C" void kernel_launch(void* const* d_in, const int* in_sizes, int n_in,
                              void* d_out, int out_size, void* d_ws, size_t ws_size,
                              hipStream_t stream) {
    (void)in_sizes; (void)n_in; (void)out_size; (void)ws_size;

    const float* x      = (const float*)d_in[0];
    const float* enc_w1 = (const float*)d_in[1];
    const float* enc_b1 = (const float*)d_in[2];
    const float* enc_w2 = (const float*)d_in[3];
    const float* enc_b2 = (const float*)d_in[4];
    const float* cb32   = (const float*)d_in[5];
    const float* dec_w1 = (const float*)d_in[6];
    const float* dec_b1 = (const float*)d_in[7];
    const float* dec_w2 = (const float*)d_in[8];
    const float* dec_b2 = (const float*)d_in[9];

    char* ws = (char*)d_ws;
    unsigned short* x_bf    = (unsigned short*)(ws + 0);        // B*D_IN bf16   32MB
    unsigned short* h_bf    = (unsigned short*)(ws + 33554432); // B*HID bf16    16MB (enc+dec reuse)
    float*          ws_ze   = (float*)(ws + 50331648);          // B*D_CODE f32  16MB
    unsigned short* zq_bf   = (unsigned short*)(ws + 67108864); // B*D_CODE bf16  8MB
    unsigned short* ws_cbbf = (unsigned short*)(ws + 75497472); // L*K*D_CODE bf16 8MB
    float*          ws_cbn  = (float*)(ws + 83886080);          // L*K f32      128KB
    unsigned short* w1t     = (unsigned short*)(ws + 84017152); // 256x512 bf16
    unsigned short* w2t     = (unsigned short*)(ws + 84279296); // 128x256 bf16
    unsigned short* d1t     = (unsigned short*)(ws + 84344832); // 256x128 bf16
    unsigned short* d2t     = (unsigned short*)(ws + 84410368); // 512x256 bf16
    float*          p_cb    = (float*)(ws + 84672512);          // 2048 f32
    float*          p_rec   = (float*)(ws + 84680704);          // 2048 f32

    float* out    = (float*)d_out;
    float* recon  = out;                                    // B*D_IN
    float* zq     = out + (size_t)B_SZ * D_IN;              // B*D_CODE
    float* codes  = zq + (size_t)B_SZ * D_CODE;             // B*LVLS
    float* scal   = codes + (size_t)B_SZ * LVLS;            // 4 scalars

    // --- prep: bf16 conversions ---
    f32_to_bf16<<<(B_SZ * D_IN) / 8 / 256, 256, 0, stream>>>(x, x_bf,
                                                             (long)B_SZ * D_IN);
    wt_prep<<<(256 * 512 + 255) / 256, 256, 0, stream>>>(enc_w1, w1t, 512, 256);
    wt_prep<<<(128 * 256 + 255) / 256, 256, 0, stream>>>(enc_w2, w2t, 256, 128);
    wt_prep<<<(256 * 128 + 255) / 256, 256, 0, stream>>>(dec_w1, d1t, 128, 256);
    wt_prep<<<(512 * 256 + 255) / 256, 256, 0, stream>>>(dec_w2, d2t, 256, 512);
    cb_prep<<<(LVLS * KCB + 255) / 256, 256, 0, stream>>>(cb32, ws_cbbf, ws_cbn);

    // --- encoder (hidden stays bf16; z_e needed in f32 for quantizer) ---
    wmma_gemm_bias<<<(2048 * 4) / 4, 128, 0, stream>>>(x_bf, w1t, enc_b1,
                                                       (float*)nullptr, h_bf,
                                                       B_SZ, 512, 256, 1);
    wmma_gemm_bias<<<(2048 * 2) / 4, 128, 0, stream>>>(h_bf, w2t, enc_b2,
                                                       ws_ze, (unsigned short*)nullptr,
                                                       B_SZ, 256, 128, 0);

    // --- residual quantization (L=4 levels fused) ---
    rq_quant<<<B_SZ / 64, 128, 0, stream>>>(ws_ze, cb32, ws_cbbf, ws_cbn,
                                            zq, zq_bf, codes, p_cb);

    // --- decoder ---
    wmma_gemm_bias<<<(2048 * 4) / 4, 128, 0, stream>>>(zq_bf, d1t, dec_b1,
                                                       (float*)nullptr, h_bf,
                                                       B_SZ, 128, 256, 1);
    wmma_gemm_bias<<<(2048 * 8) / 4, 128, 0, stream>>>(h_bf, d2t, dec_b2,
                                                       recon, (unsigned short*)nullptr,
                                                       B_SZ, 256, 512, 0);

    // --- losses ---
    recon_loss_partial<<<2048, 256, 0, stream>>>(recon, x, p_rec);
    finalize_losses<<<1, 256, 0, stream>>>(p_cb, p_rec, scal);
}
